// InteractionBlock_77232101916865
// MI455X (gfx1250) — compile-verified, matrix operand order checked by reading, not run on hardware
//
#include <hip/hip_runtime.h>
#include <math.h>

#define HIDDEN   128
#define NRBF     50
#define K1PAD    64      // GEMM1 K padded to 2 f16-WMMA steps (zeros)
#define PADW     132     // f32 LDS row stride: (132 % 64)=4 -> conflict-free column reads
#define SA_RS    72      // f16 attr row stride (halfs): 16B-aligned frags, spread banks
#define ST_RS    136     // f16 tanh-tile row stride (halfs)
#define CUTOFF   10.0f

typedef float    v2f  __attribute__((ext_vector_type(2)));
typedef float    v8f  __attribute__((ext_vector_type(8)));
typedef _Float16 v16h __attribute__((ext_vector_type(16)));

// D = A(16x4,f32) * B(4x16,f32) + C  -- fp32 matrix path (node GEMMs, exact)
__device__ __forceinline__ v8f wmma_f32_4(v2f a, v2f b, v8f c) {
  return __builtin_amdgcn_wmma_f32_16x16x4_f32(false, a, false, b, (short)0, c,
                                               false, false);
}
// D = A(16x32,f16) * B(32x16,f16) + C(f32) -- fast matrix path (filter MLP)
__device__ __forceinline__ v8f wmma_f16_32(v16h a, v16h b, v8f c) {
  return __builtin_amdgcn_wmma_f32_16x16x32_f16(false, a, false, b, (short)0, c,
                                                false, false);
}

// ISA 7.12.2, 16-bit A 16x32: element e of lane(hi) holds K = kstep*32 + pos.
// Inverse map: K -> (kstep, hi, e) so fragments are contiguous in LDS.
__device__ __forceinline__ void frag_pos(int k, int& ks, int& fhi, int& fe) {
  ks = k >> 5;
  int r = k & 31;
  fhi = (r >> 3) & 1;                 // r in [8,16)u[24,32) -> upper half-wave
  fe = (r & 7) + ((r >> 4) << 3);     // r>=16 -> elements 8..15
}

// ---------------------------------------------------------------------------
// Kernel 1: xt = x @ w_lin1^T   (also zero-init agg).  fp32 WMMA.
// ---------------------------------------------------------------------------
__global__ __launch_bounds__(256) void k_lin1(const float* __restrict__ x,
                                              const float* __restrict__ w1,
                                              float* __restrict__ xt,
                                              float* __restrict__ agg, int N) {
  __shared__ float Bw[HIDDEN * PADW];   // Bw[k*PADW+n] = w1[n][k]
  __shared__ float Sx[8][16 * PADW];

  const int tid = threadIdx.x;
  const int lane = tid & 31, wave = tid >> 5;
  const int l15 = lane & 15, hi = lane >> 4;

  for (int i = tid; i < HIDDEN * HIDDEN; i += 256) {
    int n = i >> 7, k = i & 127;
    Bw[k * PADW + n] = w1[i];
  }
  __syncthreads();

  const int ntiles = (N + 15) >> 4;
  const int tile = blockIdx.x * 8 + wave;
  if (tile >= ntiles) return;
  const int row0 = tile << 4;

  float* S = Sx[wave];
  for (int i = lane; i < 16 * HIDDEN; i += 32) {
    int r = i >> 7, c = i & 127;
    int gr = row0 + r;
    S[r * PADW + c] = (gr < N) ? x[(size_t)gr * HIDDEN + c] : 0.f;
  }

  v2f af[32];
#pragma unroll
  for (int s = 0; s < 32; ++s) {
    af[s].x = S[l15 * PADW + 4 * s + 2 * hi];
    af[s].y = S[l15 * PADW + 4 * s + 1 + 2 * hi];
  }
  for (int nt = 0; nt < 8; ++nt) {
    int n = nt * 16 + l15;
    v8f c = {};
#pragma unroll
    for (int s = 0; s < 32; ++s) {
      v2f b;
      b.x = Bw[(4 * s + 2 * hi) * PADW + n];
      b.y = Bw[(4 * s + 1 + 2 * hi) * PADW + n];
      c = wmma_f32_4(af[s], b, c);
    }
#pragma unroll
    for (int v = 0; v < 8; ++v) {
      int r = row0 + v + 8 * hi;
      if (r < N) {
        xt[(size_t)r * HIDDEN + n] = c[v];
        agg[(size_t)r * HIDDEN + n] = 0.f;
      }
    }
  }
}

// ---------------------------------------------------------------------------
// Kernel 2: fused per-edge pipeline, filter MLP on the f16 16x16x32 path.
//   W = tanh(attr@fw1+fb1)@fw2+fb2;  W *= cutoff(ew);  agg[dst] += xt[src]*W
// One wave = one 16-edge tile. Weights live in LDS in fragment-ready f16
// layout; each WMMA operand is one contiguous 32B LDS read per lane.
// ---------------------------------------------------------------------------
__global__ __launch_bounds__(256) void k_edges(
    const float* __restrict__ edge_attr, const float* __restrict__ edge_weight,
    const long long* __restrict__ srcI, const long long* __restrict__ dstI,
    const float* __restrict__ fw1, const float* __restrict__ fb1,
    const float* __restrict__ fw2, const float* __restrict__ fb2,
    const float* __restrict__ xt, float* __restrict__ agg, int E,
    int tilesPerWave) {
  // B operands, fragment-ready: idx = ((ks*2 + hi)*128 + n)*16 + e
  __shared__ _Float16 B1h[2 * 2 * HIDDEN * 16];   // fw1, K padded 50->64 (16KB)
  __shared__ _Float16 B2h[4 * 2 * HIDDEN * 16];   // fw2               (32KB)
  __shared__ float fb1s[HIDDEN], fb2s[HIDDEN];
  __shared__ _Float16 SAh[8][16 * SA_RS];         // per-wave attr tile (f16)
  __shared__ _Float16 STh[8][16 * ST_RS];         // per-wave tanh tile (f16)
  __shared__ float Scv[8][16];                    // per-wave cutoff env

  const int tid = threadIdx.x;
  const int lane = tid & 31, wave = tid >> 5;
  const int l15 = lane & 15, hi = lane >> 4;

  for (int i = tid; i < K1PAD * HIDDEN; i += 256) {
    int k = i >> 7, n = i & 127;
    float v = (k < NRBF) ? fw1[k * HIDDEN + n] : 0.f;
    int ks, fhi, fe;
    frag_pos(k, ks, fhi, fe);
    B1h[((ks * 2 + fhi) * HIDDEN + n) * 16 + fe] = (_Float16)v;
  }
  for (int i = tid; i < HIDDEN * HIDDEN; i += 256) {
    int k = i >> 7, n = i & 127;
    int ks, fhi, fe;
    frag_pos(k, ks, fhi, fe);
    B2h[((ks * 2 + fhi) * HIDDEN + n) * 16 + fe] = (_Float16)fw2[i];
  }
  if (tid < HIDDEN) {
    fb1s[tid] = fb1[tid];
    fb2s[tid] = fb2[tid];
  }
  __syncthreads();

  const int etiles = (E + 15) >> 4;
  _Float16* SA = SAh[wave];
  _Float16* ST = STh[wave];
  float* SC = Scv[wave];

  for (int it = 0; it < tilesPerWave; ++it) {
    int tile = (blockIdx.x * 8 + wave) * tilesPerWave + it;
    if (tile >= etiles) break;   // uniform across the wave
    int e0 = tile << 4;

    // prefetch next tile's edge_attr rows (16*50*4B = 3200B = 25 cachelines)
    int ntile = tile + 1;
    if (ntile < etiles && lane < 25) {
      const float* nb = edge_attr + (size_t)(ntile << 4) * NRBF;
      __builtin_prefetch(nb + lane * 32, 0, 3);
    }

    // stage edge_attr tile [16 x 50] -> f16 fragment-ready layout (K pad 64)
    for (int i = lane; i < 16 * K1PAD; i += 32) {
      int r = i >> 6, k = i & 63;
      int e = e0 + r;
      float v = (k < NRBF && e < E) ? edge_attr[(size_t)e * NRBF + k] : 0.f;
      int ks, fhi, fe;
      frag_pos(k, ks, fhi, fe);
      SA[r * SA_RS + ks * 32 + fhi * 16 + fe] = (_Float16)v;
    }
    // cosine cutoff envelope per edge
    if (lane < 16) {
      int e = e0 + lane;
      float cv = 0.f;
      if (e < E) {
        float w = edge_weight[e];
        float t = 0.5f * (__cosf(w * (3.14159265358979323846f / CUTOFF)) + 1.f);
        cv = (w < CUTOFF) ? t : 0.f;
      }
      SC[lane] = cv;
    }
    float cvv[8];
    int esrc[8], edst[8];
#pragma unroll
    for (int v = 0; v < 8; ++v) {
      int m = v + 8 * hi;
      int e = e0 + m;
      cvv[v] = SC[m];
      esrc[v] = (e < E) ? (int)srcI[e] : 0;
      edst[v] = (e < E) ? (int)dstI[e] : 0;
    }

    // GEMM1 (f16 path, K=64): t = tanh(attr @ fw1 + fb1)
    for (int nt = 0; nt < 8; ++nt) {
      int n = nt * 16 + l15;
      float bias = fb1s[n];
      v8f c = {bias, bias, bias, bias, bias, bias, bias, bias};
#pragma unroll
      for (int ks = 0; ks < 2; ++ks) {
        v16h a, b;
        __builtin_memcpy(&a, SA + l15 * SA_RS + ks * 32 + hi * 16, 32);
        __builtin_memcpy(&b, B1h + ((ks * 2 + hi) * HIDDEN + n) * 16, 32);
        c = wmma_f16_32(a, b, c);
      }
#pragma unroll
      for (int v = 0; v < 8; ++v) {
        int m = v + 8 * hi;
        int ks, fhi, fe;
        frag_pos(n, ks, fhi, fe);
        ST[m * ST_RS + ks * 32 + fhi * 16 + fe] = (_Float16)tanhf(c[v]);
      }
    }

    // GEMM2 (f16 path, K=128): W = t @ fw2 + fb2; envelope*gather*scatter
    for (int nt = 0; nt < 8; ++nt) {
      int n = nt * 16 + l15;
      float bias = fb2s[n];
      v8f c = {bias, bias, bias, bias, bias, bias, bias, bias};
#pragma unroll
      for (int ks = 0; ks < 4; ++ks) {
        v16h a, b;
        __builtin_memcpy(&a, ST + l15 * ST_RS + ks * 32 + hi * 16, 32);
        __builtin_memcpy(&b, B2h + ((ks * 2 + hi) * HIDDEN + n) * 16, 32);
        c = wmma_f16_32(a, b, c);
      }
#pragma unroll
      for (int v = 0; v < 8; ++v) {
        int e = e0 + v + 8 * hi;
        if (e < E) {
          float msg = c[v] * cvv[v] * xt[(size_t)esrc[v] * HIDDEN + n];
          unsafeAtomicAdd(&agg[(size_t)edst[v] * HIDDEN + n], msg);
        }
      }
    }
  }
}

// ---------------------------------------------------------------------------
// Kernel 3: out = tanh(agg @ w_lin2^T + b_lin2) @ w_out^T + b_out.  fp32 WMMA.
// ---------------------------------------------------------------------------
__global__ __launch_bounds__(256) void k_out(
    const float* __restrict__ agg, const float* __restrict__ w_lin2,
    const float* __restrict__ b_lin2, const float* __restrict__ w_out,
    const float* __restrict__ b_out, float* __restrict__ out, int N) {
  __shared__ float B2[HIDDEN * PADW];
  __shared__ float B3[HIDDEN * PADW];
  __shared__ float b2s[HIDDEN], b3s[HIDDEN];
  __shared__ float Sx[8][16 * PADW];

  const int tid = threadIdx.x;
  const int lane = tid & 31, wave = tid >> 5;
  const int l15 = lane & 15, hi = lane >> 4;

  for (int i = tid; i < HIDDEN * HIDDEN; i += 256) {
    int n = i >> 7, k = i & 127;
    B2[k * PADW + n] = w_lin2[i];
    B3[k * PADW + n] = w_out[i];
  }
  if (tid < HIDDEN) {
    b2s[tid] = b_lin2[tid];
    b3s[tid] = b_out[tid];
  }
  __syncthreads();

  const int ntiles = (N + 15) >> 4;
  const int tile = blockIdx.x * 8 + wave;
  if (tile >= ntiles) return;
  const int row0 = tile << 4;

  float* S = Sx[wave];
  for (int i = lane; i < 16 * HIDDEN; i += 32) {
    int r = i >> 7, c = i & 127;
    int gr = row0 + r;
    S[r * PADW + c] = (gr < N) ? agg[(size_t)gr * HIDDEN + c] : 0.f;
  }

  float hc[8][8];
  for (int nt = 0; nt < 8; ++nt) {
    int n = nt * 16 + l15;
    float bias = b2s[n];
    v8f c = {bias, bias, bias, bias, bias, bias, bias, bias};
#pragma unroll
    for (int s = 0; s < 32; ++s) {
      v2f a, b;
      a.x = S[l15 * PADW + 4 * s + 2 * hi];
      a.y = S[l15 * PADW + 4 * s + 1 + 2 * hi];
      b.x = B2[(4 * s + 2 * hi) * PADW + n];
      b.y = B2[(4 * s + 1 + 2 * hi) * PADW + n];
      c = wmma_f32_4(a, b, c);
    }
#pragma unroll
    for (int v = 0; v < 8; ++v) hc[nt][v] = tanhf(c[v]);
  }
  for (int nt = 0; nt < 8; ++nt)
#pragma unroll
    for (int v = 0; v < 8; ++v)
      S[(v + 8 * hi) * PADW + nt * 16 + l15] = hc[nt][v];

  for (int nt = 0; nt < 8; ++nt) {
    int n = nt * 16 + l15;
    float bias = b3s[n];
    v8f c = {bias, bias, bias, bias, bias, bias, bias, bias};
#pragma unroll
    for (int s = 0; s < 32; ++s) {
      v2f a, b;
      a.x = S[l15 * PADW + 4 * s + 2 * hi];
      a.y = S[l15 * PADW + 4 * s + 1 + 2 * hi];
      b.x = B3[(4 * s + 2 * hi) * PADW + n];
      b.y = B3[(4 * s + 1 + 2 * hi) * PADW + n];
      c = wmma_f32_4(a, b, c);
    }
#pragma unroll
    for (int v = 0; v < 8; ++v) {
      int r = row0 + v + 8 * hi;
      if (r < N) out[(size_t)r * HIDDEN + n] = c[v];
    }
  }
}

// ---------------------------------------------------------------------------
extern "C" void kernel_launch(void* const* d_in, const int* in_sizes, int n_in,
                              void* d_out, int out_size, void* d_ws,
                              size_t ws_size, hipStream_t stream) {
  const float* x = (const float*)d_in[0];
  const long long* eidx = (const long long*)d_in[1];  // int64 [2, E]
  const float* ew = (const float*)d_in[2];
  const float* ea = (const float*)d_in[3];
  const float* fw1 = (const float*)d_in[4];
  const float* fb1 = (const float*)d_in[5];
  const float* fw2 = (const float*)d_in[6];
  const float* fb2 = (const float*)d_in[7];
  const float* w_lin1 = (const float*)d_in[8];
  const float* w_lin2 = (const float*)d_in[9];
  const float* b_lin2 = (const float*)d_in[10];
  const float* w_out = (const float*)d_in[11];
  const float* b_out = (const float*)d_in[12];
  float* out = (float*)d_out;

  const int N = in_sizes[0] / HIDDEN;
  const int E = in_sizes[2];

  float* xt = (float*)d_ws;               // [N,128] — L2-resident
  float* agg = xt + (size_t)N * HIDDEN;   // [N,128] — L2-resident

  const int ntiles = (N + 15) / 16;
  const int nblocks = (ntiles + 7) / 8;
  k_lin1<<<nblocks, 256, 0, stream>>>(x, w_lin1, xt, agg, N);

  const int etiles = (E + 15) / 16;
  const int TPW = 4;  // edge-tiles per wave: amortize LDS weight load
  const int eblocks = (etiles + 8 * TPW - 1) / (8 * TPW);
  k_edges<<<eblocks, 256, 0, stream>>>(ea, ew, eidx, eidx + E, fw1, fb1, fw2,
                                       fb2, xt, agg, E, TPW);

  k_out<<<nblocks, 256, 0, stream>>>(agg, w_lin2, b_lin2, w_out, b_out, out, N);
}
